// GraphormerAttention_18837726560771
// MI455X (gfx1250) — matrix operand order repaired and checked by reading, not compiled, not run on hardware
//
#include <hip/hip_runtime.h>

// ---------------------------------------------------------------------------
// GraphormerAttention on gfx1250 (MI455X): WMMA f16 pipeline with
// TDM (tensor_load_to_lds) bias staging and async global->LDS weight staging.
// B=8, N=1024, D_IN=512, H=8, D_HEAD=64, INNER=512, D_OUT=512
// ---------------------------------------------------------------------------

typedef __attribute__((ext_vector_type(16))) _Float16 v16h;
typedef __attribute__((ext_vector_type(8)))  _Float16 v8h;
typedef __attribute__((ext_vector_type(4)))  _Float16 v4h;
typedef __attribute__((ext_vector_type(8)))  float    v8f;
typedef __attribute__((ext_vector_type(4)))  float    v4f;
typedef __attribute__((ext_vector_type(4)))  uint32_t u32x4;
typedef __attribute__((ext_vector_type(8)))  uint32_t u32x8;

#define NEGBIG (-1.0e30f)

__device__ __forceinline__ v8f wmma_f16(v16h a, v16h b, v8f c) {
  return __builtin_amdgcn_wmma_f32_16x16x32_f16(
      false, a, false, b, (short)0, c, false, false);
}

__device__ __forceinline__ v16h combine16(v8h a, v8h b) {
  v16h r;
#pragma unroll
  for (int i = 0; i < 8; ++i) { r[i] = a[i]; r[8 + i] = b[i]; }
  return r;
}

// A-matrix 16x32 f16: lane<16: row=lane, K=k0+0..15;
// lane>=16: row=lane-16, K=k0+16..31
__device__ __forceinline__ v16h load_A_f16(const _Float16* __restrict__ base,
                                           int ld, int row0, int k0, int lane) {
  const int l = lane & 15;
  const int hi = (lane >> 4) & 1;
  const _Float16* p = base + (size_t)(row0 + l) * ld + k0 + hi * 16;
  return combine16(*(const v8h*)p, *(const v8h*)(p + 8));
}

// B-matrix 32x16 f16 from transposed source src[col][k]; lane group = column,
// lanes<16: K=k0+0..15, lanes>=16: K=k0+16..31.
__device__ __forceinline__ v16h load_B_T(const _Float16* __restrict__ base,
                                         int ld, int col0, int k0, int lane) {
  const _Float16* p = base + (size_t)(col0 + (lane & 15)) * ld + k0 +
                      ((lane >> 4) & 1) * 16;
  return combine16(*(const v8h*)p, *(const v8h*)(p + 8));
}

// --- CDNA5 async paths ------------------------------------------------------

// Per-lane async copy of 16B global -> LDS (tracked by ASYNCcnt).
__device__ __forceinline__ void async_lds_b128(uint32_t lds_off,
                                               const void* gaddr) {
  asm volatile("global_load_async_to_lds_b128 %0, %1, off"
               :: "v"(lds_off), "v"(gaddr)
               : "memory");
}
__device__ __forceinline__ void wait_asynccnt0() {
  asm volatile("s_wait_asynccnt 0x0" ::: "memory");
}

// TDM: DMA a 2D tile (tile_dim1=16 rows x tile_dim0=256 f32, row stride 8192
// f32) from global into LDS.  D# per CDNA5 ISA.
__device__ __forceinline__ void tdm_load_bias(uint32_t lds_off,
                                              const float* gsrc) {
  const uint64_t ga = (uint64_t)(uintptr_t)gsrc;
  u32x4 g0;
  g0[0] = 1u;                                   // count=1 (valid user D#)
  g0[1] = lds_off;                              // lds_addr (bytes)
  g0[2] = (uint32_t)ga;                         // global_addr[31:0]
  g0[3] = (uint32_t)(ga >> 32) | (2u << 30);    // global_addr[56:32] | type=2
  u32x8 g1;
  g1[0] = (2u << 16);                           // wg_mask=0, data_size=4B
  g1[1] = (8192u & 0xFFFFu) << 16;              // tensor_dim0 lo16 (=8192)
  g1[2] = (1024u << 16);                        // dim0 hi=0 | tensor_dim1 lo16
  g1[3] = (256u << 16);                         // dim1 hi=0 | tile_dim0=256
  g1[4] = 16u;                                  // tile_dim1=16, tile_dim2=0
  g1[5] = 8192u;                                // tensor_dim0_stride lo32
  g1[6] = 0u;
  g1[7] = 0u;
  asm volatile("tensor_load_to_lds %0, %1" :: "s"(g0), "s"(g1) : "memory");
}

// ---------------------------------------------------------------------------
// Kernel 0a: x (f32) -> x16 (f16), same layout.
// ---------------------------------------------------------------------------
__global__ __launch_bounds__(256) void cvt_x_kernel(
    const float* __restrict__ in, _Float16* __restrict__ out) {
  const size_t i = ((size_t)blockIdx.x * 256 + threadIdx.x) * 4;
  v4f v = *(const v4f*)(in + i);
  v4h o;
#pragma unroll
  for (int j = 0; j < 4; ++j) o[j] = (_Float16)v[j];
  *(v4h*)(out + i) = o;
}

// ---------------------------------------------------------------------------
// Kernel 0b: W (K=512 x C f32, row-major) -> WT (C x 512 f16, transposed).
// ---------------------------------------------------------------------------
__global__ __launch_bounds__(256) void cvt_wT_kernel(
    const float* __restrict__ in, _Float16* __restrict__ out, int C) {
  const int idx = blockIdx.x * 256 + threadIdx.x;  // over 512*C
  const int c = idx >> 9;
  const int k = idx & 511;
  out[(size_t)c * 512 + k] = (_Float16)in[(size_t)k * C + c];
}

// ---------------------------------------------------------------------------
// Kernel 1: QKV projection.  C = x16(8192x512) @ WqkvT'(512x1536), all f16.
// Weight tiles arrive in LDS via double-buffered global_load_async_to_lds.
// Q/K stored [b,h,n,d]; V stored transposed [b,h,d,n].
// ---------------------------------------------------------------------------
__global__ __launch_bounds__(256) void qkv_proj_kernel(
    const _Float16* __restrict__ x16, const _Float16* __restrict__ WqkvT,
    _Float16* __restrict__ q_ws, _Float16* __restrict__ k_ws,
    _Float16* __restrict__ vT_ws) {
  __shared__ _Float16 WT[2][64 * 40];  // [buf][col][k], stride 40 (banks)

  const int tid = threadIdx.x;
  const int w = tid >> 5, lane = tid & 31;
  const int l = lane & 15;
  const int hi = (lane >> 4) & 1;
  const int half8 = hi * 8;
  const int m0 = blockIdx.x * 128 + w * 16;
  const int c0 = blockIdx.y * 64;  // 0..1535

  // This thread's async slice: 16B of column (c0+c), k-range part*8..+7.
  const int wc = tid >> 2, part = tid & 3;
  const _Float16* wsrc = WqkvT + (size_t)(c0 + wc) * 512 + part * 8;

  v8f acc[4] = {};

  async_lds_b128((uint32_t)(uintptr_t)&WT[0][wc * 40 + part * 8], wsrc);
  for (int ks = 0; ks < 16; ++ks) {
    const int kb = ks * 32;
    wait_asynccnt0();
    __syncthreads();
    if (ks + 1 < 16)
      async_lds_b128((uint32_t)(uintptr_t)&WT[(ks + 1) & 1][wc * 40 + part * 8],
                     wsrc + (ks + 1) * 32);

    const _Float16* WTb = &WT[ks & 1][0];
    v16h a = load_A_f16(x16, 512, m0, kb, lane);
    // Batch all B-tile LDS loads first so they clause and the 4 WMMAs issue
    // back-to-back behind a single DScnt wait.
    v16h bm[4];
#pragma unroll
    for (int t = 0; t < 4; ++t) {
      const _Float16* bp = &WTb[(t * 16 + l) * 40 + hi * 16];
      bm[t] = combine16(*(const v8h*)bp, *(const v8h*)(bp + 8));
    }
#pragma unroll
    for (int t = 0; t < 4; ++t) acc[t] = wmma_f16(a, bm[t], acc[t]);
    __syncthreads();
  }

#pragma unroll
  for (int t = 0; t < 4; ++t) {
#pragma unroll
    for (int r = 0; r < 8; ++r) {
      const int m = m0 + r + half8;
      const int bb = m >> 10, n = m & 1023;
      const int c = c0 + t * 16 + l;
      const _Float16 val = (_Float16)acc[t][r];
      if (c < 512) {
        q_ws[(((size_t)(bb * 8 + (c >> 6)) * 1024) + n) * 64 + (c & 63)] = val;
      } else if (c < 1024) {
        const int cc = c - 512;
        k_ws[(((size_t)(bb * 8 + (cc >> 6)) * 1024) + n) * 64 + (cc & 63)] = val;
      } else {
        const int cc = c - 1024;
        vT_ws[(((size_t)(bb * 8 + (cc >> 6)) * 64) + (cc & 63)) * 1024 + n] = val;
      }
    }
  }
}

// ---------------------------------------------------------------------------
// Kernel 2: flash attention; block = (b, 16-row i-tile), 8 waves = 8 heads.
// Bias tiles DMA'd by the Tensor Data Mover into double-buffered LDS.
// ---------------------------------------------------------------------------
__global__ __launch_bounds__(256) void attn_kernel(
    const _Float16* __restrict__ q_ws, const _Float16* __restrict__ k_ws,
    const _Float16* __restrict__ vT_ws, const float* __restrict__ sim_bias,
    const unsigned char* __restrict__ maskp, _Float16* __restrict__ O_ws) {
  __shared__ float biasT[2][16 * 256];      // [buf][ii][jj*8+h], 32 KB
  __shared__ _Float16 Plds[8][16 * 32];     // per-wave P tile, 8 KB

  const int tid = threadIdx.x;
  const int w = tid >> 5;                   // head
  const int lane = tid & 31;
  const int l = lane & 15;
  const int hi = (lane >> 4) & 1;
  const int half8 = hi * 8;
  const int i0 = blockIdx.x * 16;
  const int bb = blockIdx.y;

  const _Float16* qh = q_ws  + (size_t)(bb * 8 + w) * 1024 * 64;
  const _Float16* kh = k_ws  + (size_t)(bb * 8 + w) * 1024 * 64;
  const _Float16* vh = vT_ws + (size_t)(bb * 8 + w) * 64 * 1024;
  const float* bsrc0 = sim_bias + (size_t)(bb * 1024 + i0) * 1024 * 8;
  const unsigned char* mk = maskp + bb * 1024;

  const v16h aq0 = load_A_f16(qh, 64, i0, 0, lane);
  const v16h aq1 = load_A_f16(qh, 64, i0, 32, lane);

  float mrun[8], lrun[8];
  v8f O[4] = {};
#pragma unroll
  for (int r = 0; r < 8; ++r) { mrun[r] = NEGBIG; lrun[r] = 0.f; }

  _Float16* Pl = &Plds[w][0];

  // Prologue: wave 0 kicks the TDM for chunk 0 (EXEC ignored by TDM).
  if (w == 0) tdm_load_bias((uint32_t)(uintptr_t)&biasT[0][0], bsrc0);

  for (int it = 0; it < 32; ++it) {
    const int j0 = it * 32;
    if (w == 0) __builtin_amdgcn_s_wait_tensorcnt(0);
    __syncthreads();  // current buffer ready for all waves
    if (w == 0 && it + 1 < 32)
      tdm_load_bias((uint32_t)(uintptr_t)&biasT[(it + 1) & 1][0],
                    bsrc0 + (size_t)(j0 + 32) * 8);
    const float* bch = &biasT[it & 1][0];

    // --- S = Q K^T : batch all 4 K-tile loads, then 4 WMMAs ---
    v16h bk[4];
#pragma unroll
    for (int t = 0; t < 2; ++t) {
      bk[t * 2 + 0] = load_B_T(kh, 64, j0 + t * 16, 0, lane);
      bk[t * 2 + 1] = load_B_T(kh, 64, j0 + t * 16, 32, lane);
    }
    v8f S[2];
#pragma unroll
    for (int t = 0; t < 2; ++t) {
      v8f s = {};
      s = wmma_f16(aq0, bk[t * 2 + 0], s);
      s = wmma_f16(aq1, bk[t * 2 + 1], s);
      S[t] = s;
    }

    const bool msk0 = mk[j0 + l] != 0;
    const bool msk1 = mk[j0 + 16 + l] != 0;

    // --- online softmax ---
#pragma unroll
    for (int r = 0; r < 8; ++r) {
      const int ii = r + half8;
      float s0 = S[0][r] * 0.125f + bch[ii * 256 + l * 8 + w];
      float s1 = S[1][r] * 0.125f + bch[ii * 256 + (16 + l) * 8 + w];
      if (!msk0) s0 = NEGBIG;
      if (!msk1) s1 = NEGBIG;
      float mx = fmaxf(s0, s1);
#pragma unroll
      for (int off = 1; off < 16; off <<= 1)
        mx = fmaxf(mx, __shfl_xor(mx, off, 16));
      const float mnew = fmaxf(mrun[r], mx);
      const float alpha = __expf(mrun[r] - mnew);
      const float p0 = __expf(s0 - mnew);
      const float p1 = __expf(s1 - mnew);
      float ps = p0 + p1;
#pragma unroll
      for (int off = 1; off < 16; off <<= 1)
        ps += __shfl_xor(ps, off, 16);
      lrun[r] = lrun[r] * alpha + ps;
      mrun[r] = mnew;
#pragma unroll
      for (int t = 0; t < 4; ++t) O[t][r] *= alpha;
      Pl[ii * 32 + l]      = (_Float16)p0;   // C-layout -> LDS
      Pl[ii * 32 + 16 + l] = (_Float16)p1;
    }

    // --- O += P(16x32) @ V(32x64); DS ops are in-order within a wave ---
    {
      const _Float16* pp = &Pl[l * 32 + hi * 16];
      v16h ap = combine16(*(const v8h*)pp, *(const v8h*)(pp + 8));
      v16h bv[4];
#pragma unroll
      for (int t = 0; t < 4; ++t)
        bv[t] = load_B_T(vh, 1024, t * 16, j0, lane);
#pragma unroll
      for (int t = 0; t < 4; ++t) O[t] = wmma_f16(ap, bv[t], O[t]);
    }
    __syncthreads();  // all waves done with this buffer
  }

#pragma unroll
  for (int r = 0; r < 8; ++r) {
    const float inv = 1.0f / (lrun[r] + 1e-20f);
    const int n = i0 + r + half8;
    const size_t base = (size_t)(bb * 1024 + n) * 512 + w * 64;
#pragma unroll
    for (int t = 0; t < 4; ++t)
      O_ws[base + t * 16 + l] = (_Float16)(O[t][r] * inv);
  }
}

// ---------------------------------------------------------------------------
// Kernel 3: out = O(8192x512 f16) @ Wout + bout, f32 output.  Weight tiles
// via double-buffered async global->LDS.
// ---------------------------------------------------------------------------
__global__ __launch_bounds__(256) void out_proj_kernel(
    const _Float16* __restrict__ O_ws, const _Float16* __restrict__ WoutT,
    const float* __restrict__ bout, float* __restrict__ out) {
  __shared__ _Float16 WT[2][64 * 40];

  const int tid = threadIdx.x;
  const int w = tid >> 5, lane = tid & 31;
  const int l = lane & 15;
  const int hi = (lane >> 4) & 1;
  const int half8 = hi * 8;
  const int m0 = blockIdx.x * 128 + w * 16;
  const int c0 = blockIdx.y * 64;

  const int wc = tid >> 2, part = tid & 3;
  const _Float16* wsrc = WoutT + (size_t)(c0 + wc) * 512 + part * 8;

  v8f acc[4] = {};

  async_lds_b128((uint32_t)(uintptr_t)&WT[0][wc * 40 + part * 8], wsrc);
  for (int ks = 0; ks < 16; ++ks) {
    const int kb = ks * 32;
    wait_asynccnt0();
    __syncthreads();
    if (ks + 1 < 16)
      async_lds_b128((uint32_t)(uintptr_t)&WT[(ks + 1) & 1][wc * 40 + part * 8],
                     wsrc + (ks + 1) * 32);

    const _Float16* WTb = &WT[ks & 1][0];
    v16h a = load_A_f16(O_ws, 512, m0, kb, lane);
    v16h bm[4];
#pragma unroll
    for (int t = 0; t < 4; ++t) {
      const _Float16* bp = &WTb[(t * 16 + l) * 40 + hi * 16];
      bm[t] = combine16(*(const v8h*)bp, *(const v8h*)(bp + 8));
    }
#pragma unroll
    for (int t = 0; t < 4; ++t) acc[t] = wmma_f16(a, bm[t], acc[t]);
    __syncthreads();
  }

#pragma unroll
  for (int t = 0; t < 4; ++t) {
#pragma unroll
    for (int r = 0; r < 8; ++r) {
      const int m = m0 + r + half8;
      const int c = c0 + t * 16 + l;
      out[(size_t)m * 512 + c] = acc[t][r] + bout[c];
    }
  }
}

// ---------------------------------------------------------------------------
extern "C" void kernel_launch(void* const* d_in, const int* in_sizes, int n_in,
                              void* d_out, int out_size, void* d_ws,
                              size_t ws_size, hipStream_t stream) {
  (void)in_sizes; (void)n_in; (void)out_size; (void)ws_size;
  const float* x            = (const float*)d_in[0];
  const unsigned char* mask = (const unsigned char*)d_in[1];  // jax bool
  const float* sim_bias     = (const float*)d_in[2];
  const float* Wq           = (const float*)d_in[3];
  const float* Wkv          = (const float*)d_in[4];
  const float* Wout         = (const float*)d_in[5];
  const float* bout         = (const float*)d_in[6];
  float* out                = (float*)d_out;

  char* ws = (char*)d_ws;
  _Float16* q_ws   = (_Float16*)(ws + ((size_t)0));
  _Float16* k_ws   = (_Float16*)(ws + ((size_t)8 << 20));
  _Float16* vT_ws  = (_Float16*)(ws + ((size_t)16 << 20));
  _Float16* O_ws   = (_Float16*)(ws + ((size_t)24 << 20));
  _Float16* x16    = (_Float16*)(ws + ((size_t)32 << 20));
  _Float16* WqkvT  = (_Float16*)(ws + ((size_t)40 << 20));  // 1536 x 512 f16
  _Float16* WoutT  = (_Float16*)(ws + ((size_t)42 << 20));  // 512 x 512 f16

  // 0) One-time precision/layout conversion.
  cvt_x_kernel<<<4096, 256, 0, stream>>>(x, x16);                   // 8192x512
  cvt_wT_kernel<<<1024, 256, 0, stream>>>(Wq, WqkvT, 512);
  cvt_wT_kernel<<<2048, 256, 0, stream>>>(Wkv, WqkvT + (size_t)512 * 512, 1024);
  cvt_wT_kernel<<<1024, 256, 0, stream>>>(Wout, WoutT, 512);

  // 1) QKV projection: M=8192, N=1536, K=512
  qkv_proj_kernel<<<dim3(64, 24), 256, 0, stream>>>(x16, WqkvT, q_ws, k_ws,
                                                    vT_ws);
  // 2) Flash attention (TDM-staged bias)
  attn_kernel<<<dim3(64, 8), 256, 0, stream>>>(q_ws, k_ws, vT_ws, sim_bias,
                                               mask, O_ws);
  // 3) Output projection: M=8192, N=512, K=512 (+ bias)
  out_proj_kernel<<<dim3(64, 8), 256, 0, stream>>>(O_ws, WoutT, bout, out);
}
